// CellSizePredictor_43267500540540
// MI455X (gfx1250) — compile-verified, matrix operand order checked by reading, not run on hardware
//
#include <hip/hip_runtime.h>
#include <hip/hip_bf16.h>

typedef __attribute__((ext_vector_type(16))) __bf16 v16bf;
typedef __attribute__((ext_vector_type(8)))  float  v8f;

#define NN 1024
#define KTILES 32      // 1024 / 32
#define NTILES 64      // 1024 / 16
#define MT 4           // m-tiles per block
#define MROWS (MT * 16)  // 64 batch rows per block

__device__ __forceinline__ unsigned short f2bf(float f) {
    unsigned int u = __float_as_uint(f);
    u += 0x7FFFu + ((u >> 16) & 1u);       // round-to-nearest-even
    return (unsigned short)(u >> 16);
}
__device__ __forceinline__ float bf2f(unsigned short h) {
    return __uint_as_float(((unsigned int)h) << 16);
}

// ---------------------------------------------------------------------------
// Kernel 1: repack B (f32 row-major 1024x1024) -> bf16 WMMA B-fragment layout.
// Packed u16 index p = ((kt*NTILES + nt)*32 + lane)*16 + e
//   lane = g*16 + nloc ; e = r*2 + lo
//   K = kt*32 + (r>>2)*16 + g*8 + (r&3)*2 + lo ,  N = nt*16 + nloc
// ---------------------------------------------------------------------------
__global__ __launch_bounds__(256) void pack_B_bf16(const float* __restrict__ B,
                                                   unsigned short* __restrict__ Bpk) {
    int p = blockIdx.x * 256 + threadIdx.x;          // 0 .. 1048575
    int e    = p & 15;
    int lane = (p >> 4) & 31;
    int nt   = (p >> 9) & (NTILES - 1);
    int kt   = p >> 15;
    int g    = lane >> 4;
    int n    = (nt << 4) + (lane & 15);
    int r    = e >> 1, lo = e & 1;
    int k    = (kt << 5) + ((r >> 2) << 4) + (g << 3) + ((r & 3) << 1) + lo;
    Bpk[p] = f2bf(B[k * NN + n]);
}

// ---------------------------------------------------------------------------
// Kernel 2: fused  O = num*avg^alpha ;  T = O@B ;  out = rowdot(O, T + A) + C
// One block = 64 batch rows, 16 waves; wave w owns columns [w*64, w*64+64).
// Each B fragment is reused by 4 WMMAs (4 m-tiles) -> L2 B traffic = 2 GB.
// ---------------------------------------------------------------------------
__global__ __launch_bounds__(512) void fused_quadform(const float* __restrict__ x,
                                                      const float* __restrict__ A,
                                                      const unsigned short* __restrict__ Bpk,
                                                      const float* __restrict__ Cs,
                                                      const float* __restrict__ alpha,
                                                      float* __restrict__ out) {
    // A-frag layout, per m-tile: [mt][kt][lane(g,m16)][e]  -> 128 KB
    __shared__ __align__(32) unsigned short Apk[MT * KTILES * 32 * 16];
    __shared__ float Ash[NN];                                  // 4 KB
    __shared__ float red[MROWS];

    const int tid = threadIdx.x;
    const long b0 = (long)blockIdx.x * MROWS;

    for (int j = tid; j < NN; j += 512) Ash[j] = A[j];
    if (tid < MROWS) red[tid] = 0.0f;

    // ---- Stage 1: compute O rows, write packed-A layout ----
    for (int idx = tid; idx < MROWS * NN; idx += 512) {
        int m   = idx >> 10;             // 0..63
        int j   = idx & (NN - 1);
        int mt  = m >> 4;
        int m16 = m & 15;
        const float* xr = x + (b0 + m) * (2 * NN);
        float avg = xr[j];
        float num = xr[NN + j];
        float o   = num * __powf(avg, alpha[j]);
        int kt = j >> 5, ko = j & 31;
        int hh = ko >> 4;                // selects r>>2
        int k4 = ko & 15;
        int g  = k4 >> 3;                // lane half
        int rem = k4 & 7;
        int e  = ((hh << 2) + (rem >> 1)) * 2 + (rem & 1);
        Apk[((((mt * KTILES + kt) << 5) + (g << 4) + m16) << 4) + e] = f2bf(o);
    }
    __syncthreads();

    // ---- Stage 2: WMMA GEMM  T[64 x 64] per wave, K = 1024 ----
    const int wid  = tid >> 5;           // 0..15
    const int lane = tid & 31;

    v8f acc[MT][4];
#pragma unroll
    for (int mt = 0; mt < MT; ++mt)
#pragma unroll
        for (int nt = 0; nt < 4; ++nt) acc[mt][nt] = (v8f)(0.0f);

    const v16bf* __restrict__ Bvec = (const v16bf*)Bpk;  // 32B per (tile,lane)

    for (int kt = 0; kt < KTILES; ++kt) {
        if (kt + 1 < KTILES) {
            size_t pf = ((size_t)(kt + 1) * NTILES + (size_t)wid * 4) * 32 + lane;
            __builtin_prefetch((const void*)(Bvec + pf), 0, 1);
        }

        // Four A fragments (8x ds_load_b128), shared across this wave's columns
        v16bf af[MT];
#pragma unroll
        for (int mt = 0; mt < MT; ++mt)
            af[mt] = *(const v16bf*)(Apk + (((mt * KTILES + kt) * 32 + lane) << 4));

        // 4 B fragments (8x global_load_b128 clause)
        v16bf bfr[4];
        const size_t base = ((size_t)kt * NTILES + (size_t)wid * 4) * 32 + (size_t)lane;
#pragma unroll
        for (int nt = 0; nt < 4; ++nt) bfr[nt] = Bvec[base + (size_t)nt * 32];

        // 16 WMMAs: each B fragment reused 4x, each A fragment reused 4x
#pragma unroll
        for (int nt = 0; nt < 4; ++nt) {
#pragma unroll
            for (int mt = 0; mt < MT; ++mt) {
                acc[mt][nt] = __builtin_amdgcn_wmma_f32_16x16x32_bf16(
                    false, af[mt], false, bfr[nt],
                    (short)0, acc[mt][nt], false, false);
            }
        }
    }

    // ---- Stage 3: epilogue  p[m] += (t + A[j]) * O[m,j], reduce, store ----
    const int g    = lane >> 4;
    const int nloc = lane & 15;
    float p[MT][8];
#pragma unroll
    for (int mt = 0; mt < MT; ++mt)
#pragma unroll
        for (int v = 0; v < 8; ++v) p[mt][v] = 0.0f;

#pragma unroll
    for (int nt = 0; nt < 4; ++nt) {
        int j  = wid * 64 + nt * 16 + nloc;
        float aj = Ash[j];
        // inverse of the packed-A index mapping for column j
        int kt = j >> 5, ko = j & 31;
        int hh = ko >> 4;
        int k4 = ko & 15;
        int g2 = k4 >> 3;
        int rem = k4 & 7;
        int e  = ((hh << 2) + (rem >> 1)) * 2 + (rem & 1);
#pragma unroll
        for (int mt = 0; mt < MT; ++mt) {
            int tbase = (((mt * KTILES + kt) << 5) + (g2 << 4)) << 4;
#pragma unroll
            for (int v = 0; v < 8; ++v) {
                int m16 = v + (g << 3);                 // acc row within m-tile
                float ov = bf2f(Apk[tbase + (m16 << 4) + e]);
                p[mt][v] += (acc[mt][nt][v] + aj) * ov;
            }
        }
    }
    // reduce across the 16 lanes of each half (xor masks stay within halves)
#pragma unroll
    for (int off = 1; off < 16; off <<= 1) {
#pragma unroll
        for (int mt = 0; mt < MT; ++mt)
#pragma unroll
            for (int v = 0; v < 8; ++v) p[mt][v] += __shfl_xor(p[mt][v], off, 32);
    }
    if (nloc == 0) {
#pragma unroll
        for (int mt = 0; mt < MT; ++mt) {
            int mbase = mt * 16 + (g << 3);
#pragma unroll
            for (int v = 0; v < 8; ++v) atomicAdd(&red[mbase + v], p[mt][v]);
        }
    }
    __syncthreads();

    if (tid < MROWS) out[b0 + tid] = red[tid] + Cs[0];
}

extern "C" void kernel_launch(void* const* d_in, const int* in_sizes, int n_in,
                              void* d_out, int out_size, void* d_ws, size_t ws_size,
                              hipStream_t stream) {
    const float* x     = (const float*)d_in[0];
    const float* A     = (const float*)d_in[1];
    const float* B     = (const float*)d_in[2];
    const float* C     = (const float*)d_in[3];
    const float* alpha = (const float*)d_in[4];
    float* out = (float*)d_out;

    unsigned short* Bpk = (unsigned short*)d_ws;   // 2 MB bf16 packed B

    const int batch = in_sizes[0] / (2 * NN);      // 65536

    pack_B_bf16<<<(NN * NN) / 256, 256, 0, stream>>>(B, Bpk);

    fused_quadform<<<batch / MROWS, 512, 0, stream>>>(x, A, Bpk, C, alpha, out);
}
